// KronMatching_37237366456885
// MI455X (gfx1250) — matrix operand order repaired and checked by reading, not compiled, not run on hardware
//
#include <hip/hip_runtime.h>

typedef __attribute__((ext_vector_type(16))) _Float16 v16h;
typedef __attribute__((ext_vector_type(8)))  float    v8f;
typedef __attribute__((ext_vector_type(2)))  _Float16 h2;
typedef __attribute__((ext_vector_type(4)))  unsigned int u32x4;
typedef __attribute__((ext_vector_type(8)))  int i32x8;
typedef __attribute__((ext_vector_type(4)))  int i32x4;

// Problem constants (from reference: N=8, C=512, H=W=32)
constexpr int NB   = 8;
constexpr int CD   = 512;
constexpr int HD   = 32;
constexpr int WD   = 32;
constexpr int HW   = HD * WD;                 // 1024 (M and N of the GEMM)
constexpr int DD   = (2*HD-1) * (2*WD-1);     // 3969 offset bins
constexpr int BT   = 128;                     // block tile (M and N)
constexpr int KT   = 32;                      // K per step (one f16 WMMA)
constexpr int KSTEPS = CD / KT;               // 16
constexpr int ROWSTRIDE = 132;                // padded dwords per LDS row (128 + 4 TDM pad)
constexpr int TILE_F = KT * ROWSTRIDE;        // floats per staged tile (4224)

union Frag { v16h v; h2 h[8]; };

static __device__ __forceinline__ h2 pk2(float a, float b) {
    h2 r; r.x = (_Float16)a; r.y = (_Float16)b; return r;
}

// Issue one TDM load: 128x32 f32 tile (lines of 128 contiguous elements,
// 32 lines at stride 1024 elements) -> LDS with 16B pad after each 512B line.
static __device__ __forceinline__ void tdm_load_tile(const float* gsrc, const float* ldsdst) {
    unsigned long long ga = (unsigned long long)(uintptr_t)gsrc;
    unsigned lds_off = (unsigned)(unsigned long long)(uintptr_t)ldsdst; // flat addr low 32 = LDS offset

    u32x4 g0;
    g0.x = 1u;                                           // count=1, user descriptor
    g0.y = lds_off;                                      // lds_addr (bytes)
    g0.z = (unsigned)(ga & 0xffffffffu);                 // global_addr[31:0]
    g0.w = ((unsigned)(ga >> 32) & 0x01ffffffu) | (2u << 30); // addr[56:32] | type=2

    i32x8 g1;
    g1[0] = (int)((2u << 16)      // data_size = 4B
                | (1u << 20)      // pad_enable
                | (6u << 22)      // pad_interval: 128 dwords (512B) per line
                | (3u << 25));    // pad_amount: 4 dwords -> row stride 132 dwords
    g1[1] = (int)(128u << 16);    // tensor_dim0 = 128 (lo16 in dw1[31:16])
    g1[2] = (int)(32u  << 16);    // tensor_dim0 hi16 = 0 | tensor_dim1 = 32
    g1[3] = (int)(128u << 16);    // tensor_dim1 hi16 = 0 | tile_dim0 = 128
    g1[4] = 32;                   // tile_dim1 = 32 | tile_dim2 = 0
    g1[5] = (int)HW;              // tensor_dim0_stride = 1024 elements
    g1[6] = 0;                    // stride0 hi | tensor_dim1_stride lo
    g1[7] = 0;

    i32x4 g2 = {0, 0, 0, 0};
    i32x4 g3 = {0, 0, 0, 0};
    i32x8 g4 = {0, 0, 0, 0, 0, 0, 0, 0};   // extra group (clang-23 6-arg form)
    __builtin_amdgcn_tensor_load_to_lds(g0, g1, g2, g3, g4, 0);
}

__global__ __launch_bounds__(256) void corr_wmma_kernel(
        const float* __restrict__ x1, const float* __restrict__ x2,
        float* __restrict__ out)
{
    __shared__ float lds[2][2][TILE_F];   // [double buffer][A/B][32 x 132 f32]

    const int n    = blockIdx.z;
    const int p0   = blockIdx.y * BT;     // M origin (x1 spatial index)
    const int q0   = blockIdx.x * BT;     // N origin (x2 spatial index)
    const int tid  = threadIdx.x;
    const int wid  = tid >> 5;
    const int lane = tid & 31;
    const int ln   = lane & 15;
    const int hi   = lane >> 4;
    const int wm   = (wid & 3) * 32;      // wave M offset within block tile
    const int wn   = (wid >> 2) * 64;     // wave N offset within block tile

    const float* A0 = x1 + (size_t)n * CD * HW + p0;  // walk K via +k*HW
    const float* B0 = x2 + (size_t)n * CD * HW + q0;

    v8f acc[2][4] = {};

    if (wid == 0) {                        // one wave drives the TDM
        tdm_load_tile(A0, &lds[0][0][0]);
        tdm_load_tile(B0, &lds[0][1][0]);
    }

    for (int s = 0; s < KSTEPS; ++s) {
        if (wid == 0) {
            if (s + 1 < KSTEPS) {
                const float* An = A0 + (size_t)(s + 1) * KT * HW;
                const float* Bn = B0 + (size_t)(s + 1) * KT * HW;
                tdm_load_tile(An, &lds[(s + 1) & 1][0][0]);
                tdm_load_tile(Bn, &lds[(s + 1) & 1][1][0]);
                __builtin_amdgcn_s_wait_tensorcnt(2);  // stage s complete (in-order)
            } else {
                __builtin_amdgcn_s_wait_tensorcnt(0);
            }
        }
        __syncthreads();

        const float* As = &lds[s & 1][0][0];
        const float* Bs = &lds[s & 1][1][0];

        // A fragments: 16x32 f16, lane ln = row M, VGPR d holds K pair per ISA layout
        Frag fa[2];
#pragma unroll
        for (int t = 0; t < 2; ++t) {
            const int m = wm + t * 16 + ln;
#pragma unroll
            for (int d = 0; d < 8; ++d) {
                const int k = ((d < 4) ? (2 * d) : (16 + 2 * (d - 4))) + 8 * hi;
                fa[t].h[d] = pk2(As[k * ROWSTRIDE + m], As[(k + 1) * ROWSTRIDE + m]);
            }
        }
        // B fragments: 32x16 f16, lane ln = col N, VGPR d holds K pair per ISA layout (same as A)
        Frag fb[4];
#pragma unroll
        for (int t = 0; t < 4; ++t) {
            const int q = wn + t * 16 + ln;
#pragma unroll
            for (int d = 0; d < 8; ++d) {
                const int k = ((d < 4) ? (2 * d) : (16 + 2 * (d - 4))) + 8 * hi;
                fb[t].h[d] = pk2(Bs[k * ROWSTRIDE + q], Bs[(k + 1) * ROWSTRIDE + q]);
            }
        }

#pragma unroll
        for (int tm = 0; tm < 2; ++tm)
#pragma unroll
            for (int tn = 0; tn < 4; ++tn)
                acc[tm][tn] = __builtin_amdgcn_wmma_f32_16x16x32_f16(
                    false, fa[tm].v, false, fb[tn].v,
                    (short)0, acc[tm][tn], false, false);

        __syncthreads();   // everyone done with buffer s&1 before it is re-filled
    }

    // Scatter: out[n, (k-i+31)*63 + (l-j+31), i, j] = corr[p=(i,j), q=(k,l)]
    float* outn = out + (size_t)n * DD * HW;
#pragma unroll
    for (int tm = 0; tm < 2; ++tm) {
#pragma unroll
        for (int tn = 0; tn < 4; ++tn) {
            const int q  = q0 + wn + tn * 16 + ln;   // C layout: N = lane%16
            const int kq = q >> 5;
            const int l  = q & 31;
#pragma unroll
            for (int v = 0; v < 8; ++v) {
                const int p = p0 + wm + tm * 16 + v + 8 * hi;  // M = v + 8*(lane>=16)
                const int i = p >> 5;
                const int j = p & 31;
                const int d = (kq - i + (HD - 1)) * (2 * WD - 1) + (l - j + (WD - 1));
                outn[(size_t)d * HW + p] = acc[tm][tn][v];
            }
        }
    }
}

extern "C" void kernel_launch(void* const* d_in, const int* in_sizes, int n_in,
                              void* d_out, int out_size, void* d_ws, size_t ws_size,
                              hipStream_t stream) {
    (void)in_sizes; (void)n_in; (void)d_ws; (void)ws_size;
    const float* x1 = (const float*)d_in[0];
    const float* x2 = (const float*)d_in[1];
    float* out = (float*)d_out;

    // Only 1024 of the 3969 bins per (n,i,j) are written by the scatter.
    (void)hipMemsetAsync(d_out, 0, (size_t)out_size * sizeof(float), stream);

    dim3 grid(HW / BT, HW / BT, NB);   // (8, 8, 8)
    corr_wmma_kernel<<<grid, 256, 0, stream>>>(x1, x2, out);
}
